// OptimizedLSTM_1700807049567
// MI455X (gfx1250) — matrix operand order repaired and checked by reading, not compiled, not run on hardware
//
#include <hip/hip_runtime.h>

// LSTM: B=64, S=512, I=1024, H=1024.  All inputs f32.
// bf16 WMMA (v_wmma_f32_16x16x32_bf16) for the input-projection GEMM and the
// 512 sequential recurrent steps. W_hh (8MB bf16) is L2-resident (192MB L2).
// Recurrent step: K split 2-way across waves (512 waves/step, half-length
// dependent WMMA chains), LDS reduction, fused gate/cell epilogue.

#define B_ 64
#define S_ 512
#define I_ 1024
#define H_ 1024
#define G_ 4096   // 4*H
#define HH_ (H_ * H_)

typedef __attribute__((ext_vector_type(16))) __bf16 v16bf;
typedef __attribute__((ext_vector_type(8)))  __bf16 v8bf;
typedef __attribute__((ext_vector_type(8)))  float  v8f;

__device__ __forceinline__ unsigned short f32_to_bf16(float f) {
  unsigned int u = __float_as_uint(f);
  u += 0x7FFFu + ((u >> 16) & 1u);          // round-to-nearest-even
  return (unsigned short)(u >> 16);
}

// A-fragment (16x32 bf16): lane m = L&15, half kb = L>>4.
// elements 0..7 = K[base+kb*8 .. +7], elements 8..15 = K[base+16+kb*8 .. +7]
__device__ __forceinline__ v16bf load_a_frag(const unsigned short* p) {
  v8bf lo = *(const v8bf*)p;
  v8bf hi = *(const v8bf*)(p + 16);
  return __builtin_shufflevector(lo, hi, 0,1,2,3,4,5,6,7,8,9,10,11,12,13,14,15);
}

// B-fragment (32x16 bf16): lane n = L&15, half kb = L>>4; same K order as A-fragment.
__device__ __forceinline__ v16bf load_b_frag(const unsigned short* p) {
  v8bf lo = *(const v8bf*)p;
  v8bf hi = *(const v8bf*)(p + 16);
  return __builtin_shufflevector(lo, hi, 0,1,2,3,4,5,6,7,8,9,10,11,12,13,14,15);
}

__device__ __forceinline__ float fast_sigmoid(float x) {
  return 1.0f / (1.0f + __expf(-x));
}
__device__ __forceinline__ float fast_tanh(float x) {
  return 2.0f / (1.0f + __expf(-2.0f * x)) - 1.0f;
}

// ---------------- setup kernels ----------------

__global__ void __launch_bounds__(256) k_convert_x(const float* __restrict__ x,
                                                   unsigned short* __restrict__ xb) {
  long long id = (long long)blockIdx.x * 256 + threadIdx.x;   // over n/4
  float4 v = ((const float4*)x)[id];
  uint2 p;
  p.x = (unsigned int)f32_to_bf16(v.x) | ((unsigned int)f32_to_bf16(v.y) << 16);
  p.y = (unsigned int)f32_to_bf16(v.z) | ((unsigned int)f32_to_bf16(v.w) << 16);
  *(uint2*)(xb + id * 4) = p;
}

// W (K x N) f32 -> Wt (N x K) bf16   (N = 4096, K = 1024)
__global__ void __launch_bounds__(256) k_transpose_bf16(const float* __restrict__ W,
                                                        unsigned short* __restrict__ Wt) {
  int id = blockIdx.x * 256 + threadIdx.x;
  int k = id >> 12;            // id / 4096
  int n = id & 4095;           // id % 4096
  Wt[(long long)n * I_ + k] = f32_to_bf16(W[(long long)k * G_ + n]);
}

__global__ void __launch_bounds__(256) k_init_state(unsigned short* __restrict__ h0,
                                                    float* __restrict__ c) {
  int id = blockIdx.x * 256 + threadIdx.x;   // B*H = 65536
  h0[id] = 0;
  c[id] = 0.0f;
}

// ---------------- x_proj = x @ W_ih + bias ----------------
// grid = (M/16, N/512), block = 256 (8 waves); wave w -> 16 rows x 64 cols.
__global__ void __launch_bounds__(256) k_xproj_gemm(const unsigned short* __restrict__ xb,
                                                    const unsigned short* __restrict__ wt,  // (G_ x I_) bf16
                                                    const float* __restrict__ bias,
                                                    float* __restrict__ xproj) {
  const int lane  = threadIdx.x & 31;
  const int wave  = threadIdx.x >> 5;
  const int mrow0 = blockIdx.x * 16;
  const int ncol0 = blockIdx.y * 512 + wave * 64;
  const int nl = lane & 15;
  const int kb = lane >> 4;

  v8f acc[4] = {};
  const unsigned short* arow = xb + (long long)(mrow0 + nl) * I_ + kb * 8;
  const unsigned short* wrow = wt + (long long)(ncol0 + nl) * I_ + kb * 8;

  // ping-pong software pipeline over 32 k-steps
  v16bf a0 = load_a_frag(arow);
  v16bf b0[4];
  #pragma unroll
  for (int j = 0; j < 4; ++j) b0[j] = load_b_frag(wrow + j * (16 * I_));

  #pragma unroll 1
  for (int ks = 0; ks < 30; ks += 2) {
    const int k1 = (ks + 1) * 32;
    const int k2 = (ks + 2) * 32;
    v16bf a1 = load_a_frag(arow + k1);
    v16bf b1[4];
    #pragma unroll
    for (int j = 0; j < 4; ++j) b1[j] = load_b_frag(wrow + j * (16 * I_) + k1);
    #pragma unroll
    for (int j = 0; j < 4; ++j)
      acc[j] = __builtin_amdgcn_wmma_f32_16x16x32_bf16(
          false, a0, false, b0[j], (short)0, acc[j], false, false);
    a0 = load_a_frag(arow + k2);
    #pragma unroll
    for (int j = 0; j < 4; ++j) b0[j] = load_b_frag(wrow + j * (16 * I_) + k2);
    #pragma unroll
    for (int j = 0; j < 4; ++j)
      acc[j] = __builtin_amdgcn_wmma_f32_16x16x32_bf16(
          false, a1, false, b1[j], (short)0, acc[j], false, false);
  }
  {
    const int k1 = 31 * 32;
    v16bf a1 = load_a_frag(arow + k1);
    v16bf b1[4];
    #pragma unroll
    for (int j = 0; j < 4; ++j) b1[j] = load_b_frag(wrow + j * (16 * I_) + k1);
    #pragma unroll
    for (int j = 0; j < 4; ++j)
      acc[j] = __builtin_amdgcn_wmma_f32_16x16x32_bf16(
          false, a0, false, b0[j], (short)0, acc[j], false, false);
    #pragma unroll
    for (int j = 0; j < 4; ++j)
      acc[j] = __builtin_amdgcn_wmma_f32_16x16x32_bf16(
          false, a1, false, b1[j], (short)0, acc[j], false, false);
  }

  const int hi = lane >> 4;
  #pragma unroll
  for (int j = 0; j < 4; ++j) {
    const int col = ncol0 + j * 16 + nl;
    const float bj = bias[col];
    #pragma unroll
    for (int r = 0; r < 8; ++r) {
      const int row = mrow0 + r + 8 * hi;
      xproj[(long long)row * G_ + col] = acc[j][r] + bj;
    }
  }
}

// ---------------- one fused LSTM timestep ----------------
// grid = (B/16, 8), block = 512 (16 waves).
// wave = (khalf = w>>3, htile_local = w&7): each wave accumulates gates over
// half of K (512) for one (16 batch x 16 hidden) tile; khalf=1 partials are
// reduced through LDS; khalf=0 waves run the fused gate/cell epilogue.
__global__ void __launch_bounds__(512) k_lstm_step(const unsigned short* __restrict__ h_in,
                                                   unsigned short* __restrict__ h_out,
                                                   const unsigned short* __restrict__ wt_hh, // (G_ x H_) bf16
                                                   const float* __restrict__ xproj,
                                                   float* __restrict__ c,
                                                   float* __restrict__ out_hidden,
                                                   float* __restrict__ out_hT,
                                                   float* __restrict__ out_cT,
                                                   int t, int is_last) {
  __shared__ v8f red[8][4][32];                      // 32 KB partial accumulators

  const int lane  = threadIdx.x & 31;
  const int wave  = threadIdx.x >> 5;                // 0..15
  const int ht    = wave & 7;                        // local hidden tile
  const int khalf = wave >> 3;                       // 0 or 1
  const int mrow0 = blockIdx.x * 16;                 // batch tile
  const int ncol0 = (blockIdx.y * 8 + ht) * 16;      // hidden tile, 0..1023
  const int nl = lane & 15;
  const int kb = lane >> 4;

  v8f acc[4] = {};
  const unsigned short* arow = h_in + (mrow0 + nl) * H_ + khalf * 512 + kb * 8;
  // single B base; gate offsets g*HH_ (0/2/4/6 MB) fold into imm24 offsets
  const unsigned short* wrow =
      wt_hh + (long long)(ncol0 + nl) * H_ + khalf * 512 + kb * 8;

  // ping-pong software pipeline over 16 k-steps (K = 512 per wave)
  v16bf a0 = load_a_frag(arow);
  v16bf b0[4];
  #pragma unroll
  for (int g = 0; g < 4; ++g) b0[g] = load_b_frag(wrow + (long long)g * HH_);

  #pragma unroll 1
  for (int ks = 0; ks < 14; ks += 2) {
    const int k1 = (ks + 1) * 32;
    const int k2 = (ks + 2) * 32;
    // prefetch k-step ks+4 lines into near caches (global_prefetch_b8)
    __builtin_prefetch((const void*)(arow + (ks + 4) * 32), 0, 3);
    #pragma unroll
    for (int g = 0; g < 4; ++g)
      __builtin_prefetch((const void*)(wrow + (long long)g * HH_ + (ks + 4) * 32), 0, 3);

    v16bf a1 = load_a_frag(arow + k1);
    v16bf b1[4];
    #pragma unroll
    for (int g = 0; g < 4; ++g) b1[g] = load_b_frag(wrow + (long long)g * HH_ + k1);
    #pragma unroll
    for (int g = 0; g < 4; ++g)
      acc[g] = __builtin_amdgcn_wmma_f32_16x16x32_bf16(
          false, a0, false, b0[g], (short)0, acc[g], false, false);
    a0 = load_a_frag(arow + k2);
    #pragma unroll
    for (int g = 0; g < 4; ++g) b0[g] = load_b_frag(wrow + (long long)g * HH_ + k2);
    #pragma unroll
    for (int g = 0; g < 4; ++g)
      acc[g] = __builtin_amdgcn_wmma_f32_16x16x32_bf16(
          false, a1, false, b1[g], (short)0, acc[g], false, false);
  }
  {
    const int k1 = 15 * 32;
    v16bf a1 = load_a_frag(arow + k1);
    v16bf b1[4];
    #pragma unroll
    for (int g = 0; g < 4; ++g) b1[g] = load_b_frag(wrow + (long long)g * HH_ + k1);
    #pragma unroll
    for (int g = 0; g < 4; ++g)
      acc[g] = __builtin_amdgcn_wmma_f32_16x16x32_bf16(
          false, a0, false, b0[g], (short)0, acc[g], false, false);
    #pragma unroll
    for (int g = 0; g < 4; ++g)
      acc[g] = __builtin_amdgcn_wmma_f32_16x16x32_bf16(
          false, a1, false, b1[g], (short)0, acc[g], false, false);
  }

  // ---- cross-wave K reduction through LDS ----
  if (khalf == 1) {
    #pragma unroll
    for (int g = 0; g < 4; ++g) red[ht][g][lane] = acc[g];
  }
  __syncthreads();
  if (khalf == 1) return;

  #pragma unroll
  for (int g = 0; g < 4; ++g) {
    v8f p = red[ht][g][lane];
    #pragma unroll
    for (int r = 0; r < 8; ++r) acc[g][r] += p[r];
  }

  // ---- fused gate / cell epilogue (khalf == 0 waves only) ----
  const int n  = ncol0 + nl;
  const int hi = lane >> 4;
  #pragma unroll
  for (int r = 0; r < 8; ++r) {
    const int b = mrow0 + r + 8 * hi;
    const long long xbase = ((long long)b * S_ + t) * (long long)G_;
    float vi = acc[0][r] + xproj[xbase + 0 * H_ + n];
    float vf = acc[1][r] + xproj[xbase + 1 * H_ + n];
    float vg = acc[2][r] + xproj[xbase + 2 * H_ + n];
    float vo = acc[3][r] + xproj[xbase + 3 * H_ + n];
    float ig = fast_sigmoid(vi);
    float fg = fast_sigmoid(vf);
    float gg = fast_tanh(vg);
    float og = fast_sigmoid(vo);
    const int ci = b * H_ + n;
    float cn = fg * c[ci] + ig * gg;
    c[ci] = cn;
    float hn = og * fast_tanh(cn);
    out_hidden[((long long)b * S_ + t) * H_ + n] = hn;
    h_out[ci] = f32_to_bf16(hn);
    if (is_last) { out_hT[ci] = hn; out_cT[ci] = cn; }
  }
}

// ---------------- launch ----------------

extern "C" void kernel_launch(void* const* d_in, const int* in_sizes, int n_in,
                              void* d_out, int out_size, void* d_ws, size_t ws_size,
                              hipStream_t stream) {
  const float* x    = (const float*)d_in[0];
  const float* wih  = (const float*)d_in[1];
  const float* whh  = (const float*)d_in[2];
  const float* bias = (const float*)d_in[3];

  float* out        = (float*)d_out;
  float* out_hidden = out;                                   // (B,S,H)
  float* out_hT     = out + (long long)B_ * S_ * H_;         // (B,H)
  float* out_cT     = out_hT + (long long)B_ * H_;           // (B,H)

  // workspace layout (bytes)
  char* ws = (char*)d_ws;
  unsigned short* xb    = (unsigned short*)(ws);                       // 64 MB
  unsigned short* wtih  = (unsigned short*)(ws + 67108864ULL);         //  8 MB
  unsigned short* wthh  = (unsigned short*)(ws + 75497472ULL);         //  8 MB
  unsigned short* hbuf0 = (unsigned short*)(ws + 83886080ULL);         // 128 KB
  unsigned short* hbuf1 = hbuf0 + (size_t)B_ * H_;                     // 128 KB
  float*          cbuf  = (float*)(ws + 84148224ULL);                  // 256 KB
  float*          xproj = (float*)(ws + 84410368ULL);                  // 512 MB

  k_convert_x<<<(B_ * S_ * I_) / 4 / 256, 256, 0, stream>>>(x, xb);
  k_transpose_bf16<<<(I_ * G_) / 256, 256, 0, stream>>>(wih, wtih);
  k_transpose_bf16<<<(H_ * G_) / 256, 256, 0, stream>>>(whh, wthh);
  k_init_state<<<(B_ * H_) / 256, 256, 0, stream>>>(hbuf0, cbuf);

  k_xproj_gemm<<<dim3((B_ * S_) / 16, G_ / 512), 256, 0, stream>>>(xb, wtih, bias, xproj);

  for (int t = 0; t < S_; ++t) {
    const unsigned short* hin = (t & 1) ? hbuf1 : hbuf0;
    unsigned short*       hout = (t & 1) ? hbuf0 : hbuf1;
    k_lstm_step<<<dim3(B_ / 16, (H_ / 16) / 8), 512, 0, stream>>>(
        hin, hout, wthh, xproj, cbuf,
        out_hidden, out_hT, out_cT, t, (t == S_ - 1) ? 1 : 0);
  }
}
